// Sub_Clust_Cos_GLM_49478023250649
// MI455X (gfx1250) — matrix operand order repaired and checked by reading, not compiled, not run on hardware
//
#include <hip/hip_runtime.h>
#include <hip/hip_bf16.h>

#define SUB 20
#define HID 10
#define COS 24
#define TNO 200

typedef float v8f  __attribute__((ext_vector_type(8)));
typedef float v2f  __attribute__((ext_vector_type(2)));
typedef __bf16 v16bf __attribute__((ext_vector_type(16)));

// ---------------------------------------------------------------------------
// Kernel 1: stable softmax over sub axis (temp 10000 if test else 1000 w/ gate)
// writes C_syn (fp32 output) and Cw = bf16(C_syn * exp(scale)) zero-padded to
// Ep (multiple of 32) so the GEMM B-operand needs no bounds guards at all.
// ---------------------------------------------------------------------------
__global__ void softmax_cols(const float* __restrict__ raw, const float* __restrict__ g,
                             const int* __restrict__ test, const float* __restrict__ scale,
                             float* __restrict__ Cout, __bf16* __restrict__ Cw,
                             int E, int Ep) {
  int e = blockIdx.x * blockDim.x + threadIdx.x;
  if (e >= Ep) return;
  if (e >= E) {  // zero the K padding
#pragma unroll
    for (int s = 0; s < SUB; s++) Cw[s * Ep + e] = (__bf16)0.0f;
    return;
  }
  bool tst = (test[0] != 0);
  float tf = tst ? 10000.0f : 1000.0f;
  float v[SUB];
  float m = -3.4e38f;
#pragma unroll
  for (int s = 0; s < SUB; s++) {
    float x = raw[s * E + e];
    if (!tst) x += g[s * E + e];
    x *= tf;
    v[s] = x;
    m = fmaxf(m, x);
  }
  float sum = 0.0f;
#pragma unroll
  for (int s = 0; s < SUB; s++) { float ex = expf(v[s] - m); v[s] = ex; sum += ex; }
  float inv = 1.0f / sum;
  float es = expf(scale[e]);
#pragma unroll
  for (int s = 0; s < SUB; s++) {
    float c = v[s] * inv;
    Cout[s * E + e] = c;
    Cw[s * Ep + e] = (__bf16)(c * es);
  }
}

// ---------------------------------------------------------------------------
// Kernel 2: fold layer-1 weights with raised-cosine log-time basis:
// kern[src][s][hp][d] = sum_k W[s,h,k] * basis[k][d]   (hp>=HID zero-padded)
// ---------------------------------------------------------------------------
__global__ void build_kern(const float* __restrict__ We, const float* __restrict__ Wi,
                           float* __restrict__ kern) {
  int idx = blockIdx.x * blockDim.x + threadIdx.x;
  if (idx >= 2 * SUB * 16 * TNO) return;
  int d = idx % TNO;
  int rest = idx / TNO;
  int hp = rest % 16; rest /= 16;
  int s = rest % SUB;
  int src = rest / SUB;
  float out = 0.0f;
  if (hp < HID) {
    const float* W = (src == 0) ? We : Wi;
    float raw = 6.0f * logf((float)d + 1.0f + 1e-7f);
#pragma unroll
    for (int k = 0; k < COS; k++) {
      float dd = raw - 1.5707963267948966f * (float)k;
      if (dd >= -3.1415926535897931f && dd <= 3.1415926535897931f)
        out += W[(s * HID + hp) * COS + k] * (0.5f * cosf(dd) + 0.5f);
    }
  }
  kern[idx] = out;
}

// ---------------------------------------------------------------------------
// Kernel 3: syn[b,s,t] = sum_e S[b,t,e]*Cw[s,e]  via v_wmma_f32_16x16x32_bf16.
// One wave per 16-row t-tile; two N-tiles (s 0..15, 16..19) share the A frag.
// Full K-steps are completely guard-free; single guarded tail step for A.
// ---------------------------------------------------------------------------
__device__ __forceinline__ void load8f_full(v16bf& v, int off,
                                            const float* __restrict__ p, int k) {
  const float4 f0 = *(const float4*)(p + k);
  const float4 f1 = *(const float4*)(p + k + 4);
  v[off + 0] = (__bf16)f0.x; v[off + 1] = (__bf16)f0.y;
  v[off + 2] = (__bf16)f0.z; v[off + 3] = (__bf16)f0.w;
  v[off + 4] = (__bf16)f1.x; v[off + 5] = (__bf16)f1.y;
  v[off + 6] = (__bf16)f1.z; v[off + 7] = (__bf16)f1.w;
}

__device__ __forceinline__ void load8f_tail(v16bf& v, int off,
                                            const float* __restrict__ p, int k, int E) {
#pragma unroll
  for (int j = 0; j < 8; j++)
    v[off + j] = (k + j < E) ? (__bf16)p[k + j] : (__bf16)0.0f;
}

__device__ __forceinline__ v16bf loadB16(const __bf16* __restrict__ p, int k) {
  union { uint4 q[2]; v16bf v; } u;
  u.q[0] = *(const uint4*)(p + k);       // 16B aligned: row stride & k are /8
  u.q[1] = *(const uint4*)(p + k + 16);
  return u.v;
}

__global__ void gemm_syn_wmma(const float* __restrict__ S, const __bf16* __restrict__ Cw,
                              float* __restrict__ syn, int E, int Ep, int B, int T) {
  int lane = threadIdx.x & 31;
  int wv = threadIdx.x >> 5;
  int gw = blockIdx.x * (blockDim.x >> 5) + wv;
  int tilesPerB = T >> 4;
  int b = gw / tilesPerB;
  if (b >= B) return;
  int t0 = (gw % tilesPerB) << 4;
  int r16 = lane & 15;
  bool hi = (lane >= 16);

  const float* Srow = S + ((size_t)b * T + t0 + r16) * (size_t)E;  // A row: t = t0+r16
  int s0 = r16, s1 = 16 + r16;
  bool v1 = (s1 < SUB);
  const __bf16* C0 = Cw + (size_t)s0 * Ep;
  const __bf16* C1 = Cw + (size_t)(v1 ? s1 : 0) * Ep;  // invalid lanes: row 0
                                                       // (their D cols never stored)
  v8f acc0, acc1;
#pragma unroll
  for (int r = 0; r < 8; r++) { acc0[r] = 0.0f; acc1[r] = 0.0f; }

  const int kFull = E >> 5;            // guard-free steps
  const int rem = E & 31;
  const int kaOff = hi ? 8 : 0;        // bf16 A 16x32: half-split K chunks
  const int kcOff = hi ? 8 : 0;        // bf16 B 32x16: half-split K chunks

  for (int ks = 0; ks < kFull; ks++) {
    int k0 = ks << 5;
    v16bf a, bb0, bb1;
    int kb = k0 + kaOff;
    load8f_full(a, 0, Srow, kb);
    load8f_full(a, 8, Srow, kb + 16);
    int kc = k0 + kcOff;
    bb0 = loadB16(C0, kc);
    bb1 = loadB16(C1, kc);
    acc0 = __builtin_amdgcn_wmma_f32_16x16x32_bf16(false, a, false, bb0,
                                                   (short)0, acc0, false, false);
    acc1 = __builtin_amdgcn_wmma_f32_16x16x32_bf16(false, a, false, bb1,
                                                   (short)0, acc1, false, false);
  }
  if (rem) {                            // one guarded tail step (A only; B is padded)
    int k0 = kFull << 5;
    v16bf a, bb0, bb1;
    int kb = k0 + kaOff;
    load8f_tail(a, 0, Srow, kb, E);
    load8f_tail(a, 8, Srow, kb + 16, E);
    int kc = k0 + kcOff;
    bb0 = loadB16(C0, kc);
    bb1 = loadB16(C1, kc);
    acc0 = __builtin_amdgcn_wmma_f32_16x16x32_bf16(false, a, false, bb0,
                                                   (short)0, acc0, false, false);
    acc1 = __builtin_amdgcn_wmma_f32_16x16x32_bf16(false, a, false, bb1,
                                                   (short)0, acc1, false, false);
  }

  // D layout: element r -> t = t0 + r + 8*hi, col s = r16  => 8 contiguous t per lane
  int tb = t0 + (hi ? 8 : 0);
  {
    float* p = syn + ((size_t)b * SUB + s0) * (size_t)T + tb;
    float4 f0; f0.x = acc0[0]; f0.y = acc0[1]; f0.z = acc0[2]; f0.w = acc0[3];
    float4 f1; f1.x = acc0[4]; f1.y = acc0[5]; f1.z = acc0[6]; f1.w = acc0[7];
    *(float4*)p = f0; *(float4*)(p + 4) = f1;
  }
  if (v1) {
    float* p = syn + ((size_t)b * SUB + s1) * (size_t)T + tb;
    float4 f0; f0.x = acc1[0]; f0.y = acc1[1]; f0.z = acc1[2]; f0.w = acc1[3];
    float4 f1; f1.x = acc1[4]; f1.y = acc1[5]; f1.z = acc1[6]; f1.w = acc1[7];
    *(float4*)p = f0; *(float4*)(p + 4) = f1;
  }
}

// ---------------------------------------------------------------------------
// Kernel 4: causal conv + tanh + hid-contraction via v_wmma_f32_16x16x4_f32.
// D[h, t-col] += kern[h, d0..d0+3] x syn-window; fixed (b,s) per block,
// kern row-padded to 16 in LDS, per-wave 215-float syn window in LDS.
// ---------------------------------------------------------------------------
__global__ void conv_tanh_wmma(const float* __restrict__ syn_e,
                               const float* __restrict__ syn_i,
                               const float* __restrict__ kern,   // [2][SUB][16][TNO]
                               const float* __restrict__ W2,     // [SUB][HID]
                               const float* __restrict__ b1,     // [SUB*HID]
                               float* __restrict__ sub_out,      // [B][T][SUB]
                               int B, int T, int blocksPerBS) {
  __shared__ float kl[2 * 16 * TNO];
  __shared__ float win[4][2][224];

  int bs = blockIdx.x / blocksPerBS;
  int chunk = blockIdx.x % blocksPerBS;
  int b = bs / SUB, s = bs % SUB;

  for (int i = threadIdx.x; i < 2 * 16 * TNO; i += blockDim.x) {
    int src = i / (16 * TNO);
    kl[i] = kern[((size_t)src * SUB + s) * (16 * TNO) + (i % (16 * TNO))];
  }
  __syncthreads();

  int lane = threadIdx.x & 31;
  int wv = threadIdx.x >> 5;
  int nw = blockDim.x >> 5;
  int r16 = lane & 15;
  bool hi = (lane >= 16);

  float w2r[HID], br[HID];
#pragma unroll
  for (int h = 0; h < HID; h++) { w2r[h] = expf(W2[s * HID + h]); br[h] = b1[s * HID + h]; }

  const float* se = syn_e + ((size_t)b * SUB + s) * (size_t)T;
  const float* si = syn_i + ((size_t)b * SUB + s) * (size_t)T;
  int tiles = T >> 4;

  for (int tile = chunk * nw + wv; tile < tiles; tile += blocksPerBS * nw) {
    int t0 = tile << 4;
    // stage window syn[t0-199 .. t0+15] (causal zero-pad) for both sources
    for (int i = lane; i < 215; i += 32) {
      int t = t0 - 199 + i;
      win[wv][0][i] = (t >= 0) ? se[t] : 0.0f;
      win[wv][1][i] = (t >= 0) ? si[t] : 0.0f;
    }
    asm volatile("s_wait_dscnt 0" ::: "memory");

    v8f acc;
#pragma unroll
    for (int r = 0; r < 8; r++) { int h = r + (hi ? 8 : 0); acc[r] = (h < HID) ? br[h] : 0.0f; }

    for (int src = 0; src < 2; src++) {
      const float* Kr = &kl[src * 16 * TNO + r16 * TNO];  // A row h = r16
      const float* Wd = &win[wv][src][0];
      for (int d0 = 0; d0 < TNO; d0 += 4) {
        // f32 A 16x4: element j -> K = j + 2*hi ; f32 B 4x16 mirrored
        v2f a, bb;
        int ka = d0 + (hi ? 2 : 0);
        a.x = Kr[ka];
        a.y = Kr[ka + 1];
        int tb = r16 - ka + 199;           // win index of syn[t0 + r16 - ka]
        bb.x = Wd[tb];
        bb.y = Wd[tb - 1];
        acc = __builtin_amdgcn_wmma_f32_16x16x4_f32(false, a, false, bb,
                                                    (short)0, acc, false, false);
      }
    }
    // sub_out[b,t,s] = sum_h exp(W2)*tanh(pre); rows split across lane halves
    float part = 0.0f;
#pragma unroll
    for (int r = 0; r < 8; r++) {
      int h = r + (hi ? 8 : 0);
      if (h < HID) part += w2r[h] * tanhf(acc[r]);
    }
    part += __shfl_xor(part, 16, 32);
    if (!hi) sub_out[((size_t)b * T + t0 + r16) * SUB + s] = part;
    asm volatile("s_wait_dscnt 0" ::: "memory");  // WAR fence before next window fill
  }
}

// ---------------------------------------------------------------------------
// Kernel 5: final[b,t] = sum_s sub_out + V_o
// ---------------------------------------------------------------------------
__global__ void reduce_final(const float* __restrict__ sub_out, const float* __restrict__ V_o,
                             float* __restrict__ fin, int N) {
  int i = blockIdx.x * blockDim.x + threadIdx.x;
  if (i >= N) return;
  float a = V_o[0];
  const float* p = sub_out + (size_t)i * SUB;
#pragma unroll
  for (int s = 0; s < SUB; s++) a += p[s];
  fin[i] = a;
}

extern "C" void kernel_launch(void* const* d_in, const int* in_sizes, int n_in,
                              void* d_out, int out_size, void* d_ws, size_t ws_size,
                              hipStream_t stream) {
  (void)n_in; (void)out_size; (void)ws_size;
  const float* S_e     = (const float*)d_in[0];
  const float* S_i     = (const float*)d_in[1];
  const int*   test    = (const int*)d_in[3];
  const float* g_e     = (const float*)d_in[4];
  const float* g_i     = (const float*)d_in[5];
  const float* E_scale = (const float*)d_in[6];
  const float* I_scale = (const float*)d_in[7];
  const float* We1     = (const float*)d_in[8];
  const float* Wi1     = (const float*)d_in[9];
  const float* W2      = (const float*)d_in[10];
  const float* b1      = (const float*)d_in[11];
  const float* Ce_raw  = (const float*)d_in[12];
  const float* Ci_raw  = (const float*)d_in[13];
  const float* V_o     = (const float*)d_in[14];

  const int E = in_sizes[12] / SUB;         // 2000
  const int I = in_sizes[13] / SUB;         // 500
  const int B = 4;
  const int T = in_sizes[0] / (B * E);      // 20000
  const int Ep = (E + 31) & ~31;            // 2016
  const int Ip = (I + 31) & ~31;            // 512

  float* out       = (float*)d_out;
  float* final_out = out;                                   // [B*T]
  float* sub_out   = out + (size_t)B * T;                   // [B*T*SUB]
  float* Ce_out    = sub_out + (size_t)B * T * SUB;         // [SUB*E]
  float* Ci_out    = Ce_out + (size_t)SUB * E;              // [SUB*I]

  float* ws    = (float*)d_ws;
  float* syn_e = ws;                                        // [B][SUB][T]
  float* syn_i = syn_e + (size_t)B * SUB * T;               // [B][SUB][T]
  float* kern  = syn_i + (size_t)B * SUB * T;               // [2][SUB][16][TNO]
  __bf16* Cw_e = (__bf16*)(kern + (size_t)2 * SUB * 16 * TNO); // [SUB][Ep] bf16
  __bf16* Cw_i = Cw_e + (size_t)SUB * Ep;                      // [SUB][Ip] bf16

  softmax_cols<<<(Ep + 255) / 256, 256, 0, stream>>>(Ce_raw, g_e, test, E_scale,
                                                     Ce_out, Cw_e, E, Ep);
  softmax_cols<<<(Ip + 255) / 256, 256, 0, stream>>>(Ci_raw, g_i, test, I_scale,
                                                     Ci_out, Cw_i, I, Ip);
  build_kern<<<(2 * SUB * 16 * TNO + 255) / 256, 256, 0, stream>>>(We1, Wi1, kern);

  int gemmWaves = B * (T >> 4);  // one wave per 16-row t-tile
  gemm_syn_wmma<<<gemmWaves / 4, 128, 0, stream>>>(S_e, Cw_e, syn_e, E, Ep, B, T);
  gemm_syn_wmma<<<gemmWaves / 4, 128, 0, stream>>>(S_i, Cw_i, syn_i, I, Ip, B, T);

  const int blocksPerBS = 8;
  conv_tanh_wmma<<<B * SUB * blocksPerBS, 128, 0, stream>>>(syn_e, syn_i, kern, W2, b1,
                                                            sub_out, B, T, blocksPerBS);
  reduce_final<<<(B * T + 255) / 256, 256, 0, stream>>>(sub_out, V_o, final_out, B * T);
}
